// PointNet_17136919511611
// MI455X (gfx1250) — matrix operand order repaired and checked by reading, not compiled, not run on hardware
//
#include <hip/hip_runtime.h>
#include <hip/hip_bf16.h>

typedef __attribute__((ext_vector_type(16))) _Float16 v16h;
typedef __attribute__((ext_vector_type(8)))  _Float16 v8h;
typedef __attribute__((ext_vector_type(8)))  float    v8f;

#define NSEG 16384

// ---------------- WMMA fragment loaders (ISA 7.12.2 layouts, wave32) ----------
// A: 16x32 f16. lanes 0..15 hold row=lane, K in {0..7,16..23};
// lanes 16..31 hold row=lane-16, K in {8..15,24..31}. v16h element j:
//   j<8  -> k = kb + j        (kb = ks*32 + (lane>=16 ? 8 : 0))
//   j>=8 -> k = kb + 16 + (j-8)
__device__ __forceinline__ v16h ld_afrag(const _Float16* tile, int lane, int ks, int lda) {
    const int row = lane & 15;
    const int kb  = ks * 32 + ((lane >> 4) << 3);
    const _Float16* p = tile + row * lda + kb;
    v8h lo = *(const v8h*)(p);
    v8h hi = *(const v8h*)(p + 16);
    v16h a;
#pragma unroll
    for (int j = 0; j < 8; ++j) { a[j] = lo[j]; a[j + 8] = hi[j]; }
    return a;
}

// B: 32x16 f16 from LDS holding W transposed: Wt[n][k], ldb halfs.
// lanes 0..15: col=lane, K in {0..7,16..23} ; lanes 16..31: col=lane-16, K in {8..15,24..31} (within k-step).
__device__ __forceinline__ v16h ld_bfrag(const _Float16* Wt, int lane, int ct, int ks, int ldb) {
    const int n  = ct * 16 + (lane & 15);
    const int kb = ks * 32 + ((lane >> 4) << 3);
    const _Float16* p = Wt + n * ldb + kb;
    v8h lo = *(const v8h*)(p);
    v8h hi = *(const v8h*)(p + 16);
    v16h b;
#pragma unroll
    for (int j = 0; j < 8; ++j) { b[j] = lo[j]; b[j + 8] = hi[j]; }
    return b;
}

// Shared epilogue for the 64-wide layers: add bias, store f16 x, scatter-max.
// GUARD=false is the uniform full-block fast path (no per-element exec masking).
template <bool GUARD>
__device__ __forceinline__ void epilogue64(const v8f* acc, const float* __restrict__ bias,
                                           _Float16* x, float* __restrict__ segout,
                                           const int* rows, const int* soff,
                                           int colb, int Npts) {
#pragma unroll
    for (int ct = 0; ct < 4; ++ct) {
        const int col = ct * 16 + colb;
        const float bv = bias[col];
#pragma unroll
        for (int r = 0; r < 8; ++r) {
            if (!GUARD || rows[r] < Npts) {
                float y = acc[ct][r] + bv;
                x[(size_t)rows[r] * 64 + col] = (_Float16)y;
                atomicMax(segout + soff[r] + col, y);   // global_atomic_max_num_f32
            }
        }
    }
}

// ---------------- tiny utility kernels ---------------------------------------
__global__ void k_fill_f32(float* p, float v, int n) {
    int i = blockIdx.x * blockDim.x + threadIdx.x;
    if (i < n) p[i] = v;
}
__global__ void k_fixup_neginf(float* p, int n) {       // empty segments: -inf -> 0
    int i = blockIdx.x * blockDim.x + threadIdx.x;
    if (i < n) { float v = p[i]; if (v == -__builtin_inff()) p[i] = 0.0f; }
}
// W [K][Nf] f32 row-major  ->  Wt [Nf][K] f16 (so B-fragments are contiguous in k)
__global__ void k_convertWT(const float* __restrict__ W, _Float16* __restrict__ Wt, int K, int Nf) {
    int i = blockIdx.x * blockDim.x + threadIdx.x;
    if (i < K * Nf) { int k = i / Nf, n = i - k * Nf; Wt[n * K + k] = (_Float16)W[i]; }
}

// ---------------- layer 1: relu(pts@Wpos+bpos) @ W1 + b1 ----------------------
__global__ __launch_bounds__(256)
void k_layer1(const float* __restrict__ pts, const int* __restrict__ idx,
              const float* __restrict__ Wpos, const float* __restrict__ bpos,
              const _Float16* __restrict__ W1t, const float* __restrict__ b1,
              _Float16* __restrict__ x, float* __restrict__ seg, int Npts) {
    __shared__ __align__(16) _Float16 As[128 * 72];   // 128 rows x K=64 (pad 72)
    __shared__ __align__(16) _Float16 Ws[64 * 72];    // Wt1: 64 n-rows x 64 k
    __shared__ int idxs[128];
    const int t = threadIdx.x;
    const int rowBase = blockIdx.x * 128;
    const bool full = (rowBase + 128 <= Npts);

    if (t < 128) { int g = rowBase + t; idxs[t] = (full || g < Npts) ? idx[g] : 0; }
    {   // stage W1t (64x64) -> LDS, 4 threads/row, 16 halfs each
        int n = t >> 2, q = t & 3;
        const v8h* src = (const v8h*)(W1t + n * 64 + q * 16);
        v8h* dst = (v8h*)(Ws + n * 72 + q * 16);
        dst[0] = src[0]; dst[1] = src[1];
    }
    {   // h = relu(pts @ Wpos + bpos): 2 threads/point, 32 cols each (K=3 on VALU)
        int p = t >> 1, g = rowBase + p, c0 = (t & 1) * 32;
        if (full || g < Npts) {
            float p0 = pts[g * 3 + 0], p1 = pts[g * 3 + 1], p2 = pts[g * 3 + 2];
#pragma unroll
            for (int c = 0; c < 32; ++c) {
                int cc = c0 + c;
                float h = fmaf(p0, Wpos[cc], fmaf(p1, Wpos[64 + cc], fmaf(p2, Wpos[128 + cc], bpos[cc])));
                As[p * 72 + cc] = (_Float16)fmaxf(h, 0.0f);
            }
        } else {
#pragma unroll
            for (int c = 0; c < 32; ++c) As[p * 72 + c0 + c] = (_Float16)0.0f;
        }
    }
    __syncthreads();

    const int wave = t >> 5, lane = t & 31;
    const _Float16* At = As + wave * 16 * 72;
    v8f acc[4] = {};
#pragma unroll
    for (int ks = 0; ks < 2; ++ks) {                  // K=64 -> 2 k-steps
        v16h a = ld_afrag(At, lane, ks, 72);
#pragma unroll
        for (int ct = 0; ct < 4; ++ct) {              // Nout=64 -> 4 col tiles
            v16h b = ld_bfrag(Ws, lane, ct, ks, 72);
            acc[ct] = __builtin_amdgcn_wmma_f32_16x16x32_f16(false, a, false, b,
                                                             (short)0, acc[ct], false, false);
        }
    }
    const int colb = lane & 15, rsel = lane >> 4;
    int rows[8], soff[8];
#pragma unroll
    for (int r = 0; r < 8; ++r) {
        int lr = wave * 16 + r + 8 * rsel;
        rows[r] = rowBase + lr;
        soff[r] = idxs[lr] * 64;
    }
    if (full) epilogue64<false>(acc, b1, x, seg, rows, soff, colb, Npts);
    else      epilogue64<true >(acc, b1, x, seg, rows, soff, colb, Npts);
}

// ---------------- layers 2..4: relu([x | xp[idx]]) @ W + b (in-place x) -------
__global__ __launch_bounds__(256)
void k_hidden(_Float16* x, const float* __restrict__ segin, const int* __restrict__ idx,
              const _Float16* __restrict__ Wt, const float* __restrict__ bias,
              float* __restrict__ segout, int Npts) {
    __shared__ __align__(16) _Float16 As[128 * 136];  // 128 rows x K=128 (pad 136)
    __shared__ __align__(16) _Float16 Ws[64 * 136];   // Wt: 64 n-rows x 128 k
    __shared__ int idxs[128];
    const int t = threadIdx.x;
    const int rowBase = blockIdx.x * 128;
    const bool full = (rowBase + 128 <= Npts);

    if (t < 128) { int g = rowBase + t; idxs[t] = (full || g < Npts) ? idx[g] : 0; }
    {   // stage Wt (64x128) -> LDS, 4 threads/row, 32 halfs each
        int n = t >> 2, q = t & 3;
        const v8h* src = (const v8h*)(Wt + n * 128 + q * 32);
        v8h* dst = (v8h*)(Ws + n * 136 + q * 32);
        dst[0] = src[0]; dst[1] = src[1]; dst[2] = src[2]; dst[3] = src[3];
    }
    {   // A panel: k<64 = relu(x[row]) f16 ; k>=64 = relu(segin[idx[row]]) f32->f16
        int r = t >> 1, g = rowBase + r, half = t & 1;
        _Float16* d = As + r * 136 + half * 64;
        if (half == 0) {
            if (full || g < Npts) {
                const v8h* s = (const v8h*)(x + (size_t)g * 64);
                _Float16 hz = (_Float16)0.0f;
#pragma unroll
                for (int q = 0; q < 8; ++q) {
                    v8h v = s[q];
#pragma unroll
                    for (int j = 0; j < 8; ++j) v[j] = (v[j] > hz) ? v[j] : hz;
                    *(v8h*)(d + q * 8) = v;
                }
            } else { for (int q = 0; q < 64; ++q) d[q] = (_Float16)0.0f; }
        } else {
            if (full || g < Npts) {
                int s = idx[g];
                const float4* sp = (const float4*)(segin + (size_t)s * 64);
#pragma unroll
                for (int q = 0; q < 16; ++q) {
                    float4 v = sp[q];
                    d[q * 4 + 0] = (_Float16)fmaxf(v.x, 0.0f);
                    d[q * 4 + 1] = (_Float16)fmaxf(v.y, 0.0f);
                    d[q * 4 + 2] = (_Float16)fmaxf(v.z, 0.0f);
                    d[q * 4 + 3] = (_Float16)fmaxf(v.w, 0.0f);
                }
            } else { for (int q = 0; q < 64; ++q) d[q] = (_Float16)0.0f; }
        }
    }
    __syncthreads();

    const int wave = t >> 5, lane = t & 31;
    const _Float16* At = As + wave * 16 * 136;
    v8f acc[4] = {};
#pragma unroll
    for (int ks = 0; ks < 4; ++ks) {                  // K=128 -> 4 k-steps
        v16h a = ld_afrag(At, lane, ks, 136);
#pragma unroll
        for (int ct = 0; ct < 4; ++ct) {
            v16h b = ld_bfrag(Ws, lane, ct, ks, 136);
            acc[ct] = __builtin_amdgcn_wmma_f32_16x16x32_f16(false, a, false, b,
                                                             (short)0, acc[ct], false, false);
        }
    }
    const int colb = lane & 15, rsel = lane >> 4;
    int rows[8], soff[8];
#pragma unroll
    for (int r = 0; r < 8; ++r) {
        int lr = wave * 16 + r + 8 * rsel;
        rows[r] = rowBase + lr;
        soff[r] = idxs[lr] * 64;
    }
    if (full) epilogue64<false>(acc, bias, x, segout, rows, soff, colb, Npts);
    else      epilogue64<true >(acc, bias, x, segout, rows, soff, colb, Npts);
}

// ---------------- head: relu(xp4) @ Wout + bout -> [16384,128] ----------------
__global__ __launch_bounds__(256)
void k_final(const float* __restrict__ segin, const _Float16* __restrict__ Wt, // [128][64]
             const float* __restrict__ bias, float* __restrict__ out) {
    __shared__ __align__(16) _Float16 As[128 * 72];   // 128 seg-rows x K=64
    __shared__ __align__(16) _Float16 Ws[128 * 72];   // Wt: 128 n-rows x 64 k
    const int t = threadIdx.x;
    const int rowBase = blockIdx.x * 128;
    {   // stage Wt (128x64): 2 threads/row, 32 halfs each
        int n = t >> 1, q = t & 1;
        const v8h* src = (const v8h*)(Wt + n * 64 + q * 32);
        v8h* dst = (v8h*)(Ws + n * 72 + q * 32);
        dst[0] = src[0]; dst[1] = src[1]; dst[2] = src[2]; dst[3] = src[3];
    }
    {   // A = relu(segin) f32 -> f16
        int r = t >> 1, q = t & 1, g = rowBase + r;
        const float4* sp = (const float4*)(segin + (size_t)g * 64 + q * 32);
        _Float16* d = As + r * 72 + q * 32;
#pragma unroll
        for (int i = 0; i < 8; ++i) {
            float4 v = sp[i];
            d[i * 4 + 0] = (_Float16)fmaxf(v.x, 0.0f);
            d[i * 4 + 1] = (_Float16)fmaxf(v.y, 0.0f);
            d[i * 4 + 2] = (_Float16)fmaxf(v.z, 0.0f);
            d[i * 4 + 3] = (_Float16)fmaxf(v.w, 0.0f);
        }
    }
    __syncthreads();

    const int wave = t >> 5, lane = t & 31;
    const _Float16* At = As + wave * 16 * 72;
    v8f acc[8] = {};
#pragma unroll
    for (int ks = 0; ks < 2; ++ks) {                  // K=64
        v16h a = ld_afrag(At, lane, ks, 72);
#pragma unroll
        for (int ct = 0; ct < 8; ++ct) {              // Nout=128 -> 8 col tiles
            v16h b = ld_bfrag(Ws, lane, ct, ks, 72);
            acc[ct] = __builtin_amdgcn_wmma_f32_16x16x32_f16(false, a, false, b,
                                                             (short)0, acc[ct], false, false);
        }
    }
    const int colb = lane & 15, rsel = lane >> 4;
#pragma unroll
    for (int ct = 0; ct < 8; ++ct) {
        int col = ct * 16 + colb;
        float bv = bias[col];
#pragma unroll
        for (int r = 0; r < 8; ++r) {
            int g = rowBase + wave * 16 + r + 8 * rsel;
            out[(size_t)g * 128 + col] = acc[ct][r] + bv;
        }
    }
}

// ---------------- host side ---------------------------------------------------
extern "C" void kernel_launch(void* const* d_in, const int* in_sizes, int n_in,
                              void* d_out, int out_size, void* d_ws, size_t ws_size,
                              hipStream_t stream) {
    (void)n_in; (void)out_size; (void)ws_size;
    const float* pts  = (const float*)d_in[0];
    const int*   idx  = (const int*)d_in[1];
    const float* Wpos = (const float*)d_in[3];
    const float* bpos = (const float*)d_in[4];
    const float* W1   = (const float*)d_in[5];  const float* b1 = (const float*)d_in[6];
    const float* W2   = (const float*)d_in[7];  const float* b2 = (const float*)d_in[8];
    const float* W3   = (const float*)d_in[9];  const float* b3 = (const float*)d_in[10];
    const float* W4   = (const float*)d_in[11]; const float* b4 = (const float*)d_in[12];
    const float* Wout = (const float*)d_in[13]; const float* bout = (const float*)d_in[14];
    float* out = (float*)d_out;
    const int Npts = in_sizes[0] / 3;

    // workspace carve-out (deterministic layout every call)
    char* ws = (char*)d_ws;
    size_t off = 0;
    auto carve = [&](size_t bytes) { char* p = ws + off; off = (off + bytes + 255) & ~(size_t)255; return p; };
    _Float16* x    = (_Float16*)carve((size_t)Npts * 64 * 2);   // f16 activations [N,64]
    float*    segA = (float*)carve((size_t)NSEG * 64 * 4);
    float*    segB = (float*)carve((size_t)NSEG * 64 * 4);
    _Float16* W1t  = (_Float16*)carve(64 * 64 * 2);
    _Float16* W2t  = (_Float16*)carve(64 * 128 * 2);
    _Float16* W3t  = (_Float16*)carve(64 * 128 * 2);
    _Float16* W4t  = (_Float16*)carve(64 * 128 * 2);
    _Float16* Wot  = (_Float16*)carve(128 * 64 * 2);

    // f16 transposed weights
    k_convertWT<<<(64 * 64 + 255) / 256, 256, 0, stream>>>(W1, W1t, 64, 64);
    k_convertWT<<<(128 * 64 + 255) / 256, 256, 0, stream>>>(W2, W2t, 128, 64);
    k_convertWT<<<(128 * 64 + 255) / 256, 256, 0, stream>>>(W3, W3t, 128, 64);
    k_convertWT<<<(128 * 64 + 255) / 256, 256, 0, stream>>>(W4, W4t, 128, 64);
    k_convertWT<<<(64 * 128 + 255) / 256, 256, 0, stream>>>(Wout, Wot, 64, 128);

    const int segN = NSEG * 64;
    const int segGrid = (segN + 255) / 256;
    const int nb = (Npts + 127) / 128;

    k_fill_f32<<<segGrid, 256, 0, stream>>>(segA, -__builtin_inff(), segN);
    k_layer1<<<nb, 256, 0, stream>>>(pts, idx, Wpos, bpos, W1t, b1, x, segA, Npts);
    k_fixup_neginf<<<segGrid, 256, 0, stream>>>(segA, segN);

    k_fill_f32<<<segGrid, 256, 0, stream>>>(segB, -__builtin_inff(), segN);
    k_hidden<<<nb, 256, 0, stream>>>(x, segA, idx, W2t, b2, segB, Npts);
    k_fixup_neginf<<<segGrid, 256, 0, stream>>>(segB, segN);

    k_fill_f32<<<segGrid, 256, 0, stream>>>(segA, -__builtin_inff(), segN);
    k_hidden<<<nb, 256, 0, stream>>>(x, segB, idx, W3t, b3, segA, Npts);
    k_fixup_neginf<<<segGrid, 256, 0, stream>>>(segA, segN);

    k_fill_f32<<<segGrid, 256, 0, stream>>>(segB, -__builtin_inff(), segN);
    k_hidden<<<nb, 256, 0, stream>>>(x, segA, idx, W4t, b4, segB, Npts);
    k_fixup_neginf<<<segGrid, 256, 0, stream>>>(segB, segN);

    k_final<<<NSEG / 128, 256, 0, stream>>>(segB, Wot, bout, out);
}
